// myNetwork_8641474200055
// MI455X (gfx1250) — matrix operand order repaired and checked by reading, not compiled, not run on hardware
//
#include <hip/hip_runtime.h>

// ---------------- problem constants ----------------
constexpr int BB = 32;      // batch (graphs)
constexpr int NN = 1024;    // nodes
constexpr int NF = 64;      // input features
constexpr int HH = 128;     // hidden
constexpr int LL = 2;       // GIN layers
constexpr int KK = 12;      // shapelets per graph
constexpr int TAR = 288;
constexpr float BN_EPS = 1e-5f;

typedef __attribute__((ext_vector_type(16))) __bf16 v16bf;
typedef __attribute__((ext_vector_type(8)))  float  v8f;

__device__ inline v8f vzero8() {
    v8f v = {0.f,0.f,0.f,0.f,0.f,0.f,0.f,0.f};
    return v;
}

// ---- A fragment 16x32 (MxK), ISA 7.12.2 layout, via 4x b128 loads ----
// lanes 0-15: row M=lane, K in {0..7,16..23}; lanes 16-31: row M=lane-16, K in {8..15,24..31}
__device__ inline v16bf load_a_frag(const float* __restrict__ A, int lda, int m0, int kb) {
    int lane = threadIdx.x & 31;
    const float* row = A + (size_t)(m0 + (lane & 15)) * lda + kb + ((lane >> 4) << 3);
    float4 p0 = *(const float4*)(row + 0);
    float4 p1 = *(const float4*)(row + 4);
    float4 p2 = *(const float4*)(row + 16);
    float4 p3 = *(const float4*)(row + 20);
    v16bf a;
    a[0]=(__bf16)p0.x;  a[1]=(__bf16)p0.y;  a[2]=(__bf16)p0.z;  a[3]=(__bf16)p0.w;
    a[4]=(__bf16)p1.x;  a[5]=(__bf16)p1.y;  a[6]=(__bf16)p1.z;  a[7]=(__bf16)p1.w;
    a[8]=(__bf16)p2.x;  a[9]=(__bf16)p2.y;  a[10]=(__bf16)p2.z; a[11]=(__bf16)p2.w;
    a[12]=(__bf16)p3.x; a[13]=(__bf16)p3.y; a[14]=(__bf16)p3.z; a[15]=(__bf16)p3.w;
    return a;
}

// ---- cooperative staging of a 32-row x 128-col f32 B panel into swizzled bf16 LDS ----
// LDS layout (bf16): dst[n*32 + p], p = 16*((k>>3)&1) + (k&7) + 8*(k>>4) (ISA table order), so
// fragment (n, khalf) = 32 contiguous bytes at dst + n*32 + khalf*16.
// blockDim.x must be 256. Coalesced: adjacent threads load adjacent n.
__device__ inline void stage_b_panel(__bf16* dst, const float* __restrict__ Bm,
                                     int ldb, int kb, int n0) {
    int tid = threadIdx.x;
#pragma unroll
    for (int j = 0; j < 8; ++j) {
        int idx = j * 256 + tid;            // 2048 even-k pair slots
        int n = idx & 127;
        int k = (idx >> 7) << 1;            // even k in [0,32)
        float f0 = Bm[(size_t)(kb + k)     * ldb + n0 + n];
        float f1 = Bm[(size_t)(kb + k + 1) * ldb + n0 + n];
        union { __bf16 h[2]; unsigned u; } p;
        p.h[0] = (__bf16)f0; p.h[1] = (__bf16)f1;
        *reinterpret_cast<unsigned*>(dst + n * 32 + (((k >> 3) & 1) << 4) + (k & 7) + ((k >> 4) << 3)) = p.u;   // 4B-aligned ds_store_b32
    }
}

// ---- B fragment from swizzled LDS panel: one 32B read (2x ds_load_b128) ----
__device__ inline v16bf frag_from_lds(const __bf16* panel, int nt) {
    int lane = threadIdx.x & 31;
    return ((const v16bf*)panel)[(nt * 16 + (lane & 15)) * 2 + (lane >> 4)];
}

// ---- f32 C/D 16x16: VGPR v holds row (v + 8*(lane>=16)), col = lane&15 ----
// mode 0: bias+relu ; mode 1: bias+bn+relu ; mode 2: bias+bn (no relu)
__device__ inline void store_tile(float* __restrict__ D, int ldd, int m0, int n0, v8f acc,
                                  const float* __restrict__ bias,
                                  const float* __restrict__ g, const float* __restrict__ be,
                                  const float* __restrict__ mu, const float* __restrict__ var,
                                  int mode) {
    int lane = threadIdx.x & 31;
    int n = n0 + (lane & 15);
    float bi = bias[n];
    float scale = 1.f, shift = 0.f;
    if (mode >= 1) {
        float rs = rsqrtf(var[n] + BN_EPS) * g[n];
        scale = rs;
        shift = be[n] - mu[n] * rs;
    }
    int mh = (lane >> 4) << 3;
#pragma unroll
    for (int v = 0; v < 8; ++v) {
        float val = acc[v] + bi;
        if (mode >= 1) val = val * scale + shift;
        if (mode != 2) val = fmaxf(val, 0.f);
        D[(size_t)(m0 + mh + v) * ldd + n] = val;
    }
}

// ============ generic row-tile GEMM: D[M,ldb] = epilogue(A[M,K] @ B[K,ldb]) ============
// one wave -> 16 rows x 128 cols (cols n0 = blockIdx.y*128 .. +128); 8 waves per block.
__global__ void __launch_bounds__(256) gemm_rowtile(
    const float* __restrict__ A, const float* __restrict__ Bm, float* __restrict__ D,
    int M, int K, int ldb,
    const float* __restrict__ bias,
    const float* __restrict__ g, const float* __restrict__ be,
    const float* __restrict__ mu, const float* __restrict__ var, int mode) {
    __shared__ __align__(32) __bf16 ldsB[128 * 128];   // up to 4 swizzled panels = 32 KB
    int wid = blockIdx.x * 8 + (threadIdx.x >> 5);
    int m0 = wid * 16;
    bool active = (m0 < M);          // wave-uniform; must NOT return early (barriers)
    int n0 = blockIdx.y * 128;
    v8f acc[8];
#pragma unroll
    for (int nt = 0; nt < 8; ++nt) acc[nt] = vzero8();

    for (int kc = 0; kc < K; kc += 128) {
        int kcend = K - kc; if (kcend > 128) kcend = 128;   // always a multiple of 32 here
        for (int kb = 0; kb < kcend; kb += 32)
            stage_b_panel(ldsB + kb * 128, Bm, ldb, kc + kb, n0);
        __syncthreads();
        if (active) {
            for (int kb = 0; kb < kcend; kb += 32) {
                v16bf a = load_a_frag(A, K, m0, kc + kb);
                const __bf16* panel = ldsB + kb * 128;
                v16bf bf[8];
#pragma unroll
                for (int nt = 0; nt < 8; ++nt) bf[nt] = frag_from_lds(panel, nt);
#pragma unroll
                for (int nt = 0; nt < 8; ++nt)
                    acc[nt] = __builtin_amdgcn_wmma_f32_16x16x32_bf16(
                        false, a, false, bf[nt], (short)0, acc[nt], false, false);
            }
        }
        __syncthreads();
    }
    if (active) {
#pragma unroll
        for (int nt = 0; nt < 8; ++nt)
            store_tile(D, ldb, m0, n0 + nt * 16, acc[nt], bias, g, be, mu, var, mode);
    }
}

// ============ z = xh + mask^T @ xh  (per graph), mask = (adj>0) ============
// A[i,j] = (adj[g][j][i] > 0), built on the fly (0/1 exact in bf16); adj read exactly once,
// coalesced across lanes (adjacent lanes -> adjacent i). xh panels staged via LDS.
__global__ void __launch_bounds__(256) gin_aggregate(
    const float* __restrict__ adj, const float* __restrict__ xh, float* __restrict__ z) {
    __shared__ __align__(32) __bf16 ldsB[128 * 128];
    int wid = blockIdx.x * 8 + (threadIdx.x >> 5);
    int gph = wid >> 6;                 // 64 row tiles per graph; whole block = same graph
    int m0  = (wid & 63) * 16;
    const float* adjg = adj + (size_t)gph * NN * NN;
    const float* xg   = xh  + (size_t)gph * NN * HH;
    float*       zg   = z   + (size_t)gph * NN * HH;

    int lane   = threadIdx.x & 31;
    int m      = m0 + (lane & 15);
    int khalf8 = (lane >> 4) << 3;

    v8f acc[8];
#pragma unroll
    for (int nt = 0; nt < 8; ++nt) acc[nt] = vzero8();

    for (int kc = 0; kc < NN; kc += 128) {
        for (int kb = 0; kb < 128; kb += 32)
            stage_b_panel(ldsB + kb * 128, xg, HH, kc + kb, 0);
        __syncthreads();
        for (int kb = 0; kb < 128; kb += 32) {
            v16bf a;
#pragma unroll
            for (int i = 0; i < 16; ++i) {
                int k = kc + kb + ((i & 8) << 1) + khalf8 + (i & 7);
                a[i] = (__bf16)((adjg[(size_t)k * NN + m] > 0.f) ? 1.f : 0.f);
            }
            const __bf16* panel = ldsB + kb * 128;
            v16bf bf[8];
#pragma unroll
            for (int nt = 0; nt < 8; ++nt) bf[nt] = frag_from_lds(panel, nt);
#pragma unroll
            for (int nt = 0; nt < 8; ++nt)
                acc[nt] = __builtin_amdgcn_wmma_f32_16x16x32_bf16(
                    false, a, false, bf[nt], (short)0, acc[nt], false, false);
        }
        __syncthreads();
    }
    int mh = (lane >> 4) << 3;
#pragma unroll
    for (int nt = 0; nt < 8; ++nt) {
        int n = nt * 16 + (lane & 15);
#pragma unroll
        for (int v = 0; v < 8; ++v) {
            size_t off = (size_t)(m0 + mh + v) * HH + n;
            zg[off] = acc[nt][v] + xg[off];     // fused z = xh + agg
        }
    }
}

// ============ per-(graph,layer) column sums over nodes: sums[b, l*128+h] ============
__global__ void colsum_kernel(const float* __restrict__ h0, const float* __restrict__ h1,
                              float* __restrict__ sums) {
    int b = blockIdx.x, l = blockIdx.y, h = threadIdx.x;   // 128 threads
    const float* src = (l == 0 ? h0 : h1) + (size_t)b * NN * HH + h;
    float s = 0.f;
    for (int n = 0; n < NN; ++n) s += src[(size_t)n * HH];
    sums[b * (LL * HH) + l * HH + h] = s;
}

// ============ graph_Re[b,h] = relu(sums[b,:256] @ mlp2_W + b) ============
__global__ void graph_re_kernel(const float* __restrict__ sums, const float* __restrict__ W,
                                const float* __restrict__ bias, float* __restrict__ gre) {
    int b = blockIdx.x, h = threadIdx.x;    // 128 threads
    float s = bias[h];
    const float* sb = sums + b * (LL * HH);
    for (int j = 0; j < LL * HH; ++j) s += sb[j] * W[j * HH + h];
    gre[b * HH + h] = fmaxf(s, 0.f);
}

// ============ dis + stable top-K (matches lax.top_k tie-break: lowest index) ============
__global__ void topk_kernel(const float* __restrict__ xh, const float* __restrict__ h1,
                            const float* __restrict__ gre, int* __restrict__ idxout) {
    __shared__ float dis[NN];
    __shared__ float gr[HH];
    __shared__ float rv[256];
    __shared__ int   ri[256];
    int b = blockIdx.x, t = threadIdx.x;    // 256 threads
    const float* xg = xh + (size_t)b * NN * HH;
    const float* hg = h1 + (size_t)b * NN * HH;
    if (t < HH) gr[t] = gre[b * HH + t];
    __syncthreads();
    for (int n = t; n < NN; n += 256) {
        float s = 0.f;
        const float* xr = xg + (size_t)n * HH;
        const float* hr = hg + (size_t)n * HH;
        for (int h = 0; h < HH; ++h) s += (xr[h] + hr[h]) * gr[h];
        dis[n] = -s;
    }
    __syncthreads();
    for (int kk = 0; kk < KK; ++kk) {
        float best = -3.402823466e38f; int besti = NN;
        for (int n = t; n < NN; n += 256) {
            float v = dis[n];
            if (v > best || (v == best && n < besti)) { best = v; besti = n; }
        }
        rv[t] = best; ri[t] = besti;
        __syncthreads();
        for (int off = 128; off > 0; off >>= 1) {
            if (t < off) {
                if (rv[t + off] > rv[t] || (rv[t + off] == rv[t] && ri[t + off] < ri[t])) {
                    rv[t] = rv[t + off]; ri[t] = ri[t + off];
                }
            }
            __syncthreads();
        }
        if (t == 0) { idxout[b * KK + kk] = ri[0]; dis[ri[0]] = -3.402823466e38f; }
        __syncthreads();
    }
}

// ============ tail layer 1: hrow[b, j*512+c] = relu(emb(idx) @ net_W1 + b1) ============
__global__ void tail1_kernel(const int* __restrict__ idx, const float* __restrict__ emb,
                             const float* __restrict__ W1, const float* __restrict__ b1,
                             float* __restrict__ hrow) {
    int gid = blockIdx.x * blockDim.x + threadIdx.x;
    if (gid >= BB * KK * 512) return;
    int b = gid / (KK * 512);
    int r = gid - b * (KK * 512);
    int j = r >> 9, c = r & 511;
    int id = idx[b * KK + j];
    float e0 = emb[id * 2], e1 = emb[id * 2 + 1];
    float v = e0 * W1[c] + e1 * W1[512 + c] + b1[c];
    hrow[gid] = fmaxf(v, 0.f);
}

// ============ final: out[b,o] = a2[b,:] @ net_W3 + b3 ============
__global__ void tail3_kernel(const float* __restrict__ a2, const float* __restrict__ W3,
                             const float* __restrict__ b3, float* __restrict__ out) {
    int gid = blockIdx.x * blockDim.x + threadIdx.x;
    if (gid >= BB * TAR) return;
    int b = gid / TAR, o = gid - b * TAR;
    float s = b3[o];
    const float* ar = a2 + b * 512;
    for (int c = 0; c < 512; ++c) s += ar[c] * W3[c * TAR + o];
    out[gid] = s;
}

// ================================================================
extern "C" void kernel_launch(void* const* d_in, const int* in_sizes, int n_in,
                              void* d_out, int out_size, void* d_ws, size_t ws_size,
                              hipStream_t stream) {
    const float* x          = (const float*)d_in[0];
    const float* adj        = (const float*)d_in[1];
    const float* mlp1_W     = (const float*)d_in[2];
    const float* mlp1_b     = (const float*)d_in[3];
    const float* gin_lin1_W = (const float*)d_in[4];
    const float* gin_lin1_b = (const float*)d_in[5];
    const float* gin_bn1_g  = (const float*)d_in[6];
    const float* gin_bn1_b  = (const float*)d_in[7];
    const float* gin_bn1_m  = (const float*)d_in[8];
    const float* gin_bn1_v  = (const float*)d_in[9];
    const float* gin_lin2_W = (const float*)d_in[10];
    const float* gin_lin2_b = (const float*)d_in[11];
    const float* gin_bn2_g  = (const float*)d_in[12];
    const float* gin_bn2_b  = (const float*)d_in[13];
    const float* gin_bn2_m  = (const float*)d_in[14];
    const float* gin_bn2_v  = (const float*)d_in[15];
    const float* mlp2_W     = (const float*)d_in[16];
    const float* mlp2_b     = (const float*)d_in[17];
    const float* emb_table  = (const float*)d_in[18];
    const float* net_W1     = (const float*)d_in[19];
    const float* net_b1     = (const float*)d_in[20];
    const float* net_W2     = (const float*)d_in[21];
    const float* net_b2     = (const float*)d_in[22];
    const float* net_bn_g   = (const float*)d_in[23];
    const float* net_bn_b   = (const float*)d_in[24];
    const float* net_bn_m   = (const float*)d_in[25];
    const float* net_bn_v   = (const float*)d_in[26];
    const float* net_W3     = (const float*)d_in[27];
    const float* net_b3     = (const float*)d_in[28];

    // workspace layout (floats)
    const size_t NB = (size_t)BB * NN * HH;       // 4,194,304
    float* ws   = (float*)d_ws;
    float* xh   = ws;
    float* z    = xh + NB;
    float* t    = z  + NB;
    float* h0   = t  + NB;
    float* h1   = h0 + NB;
    float* sums = h1 + NB;                        // 32*256
    float* gre  = sums + BB * LL * HH;            // 32*128
    float* hrow = gre  + BB * HH;                 // 32*6144
    float* a2   = hrow + (size_t)BB * KK * 512;   // 32*512
    int*   idx  = (int*)(a2 + BB * 512);          // 32*12

    const int M = BB * NN;              // 32768 rows, 2048 tiles, 256 blocks of 8 waves

    // xh = relu(x @ mlp1_W + b)
    gemm_rowtile<<<dim3(256, 1), 256, 0, stream>>>(
        x, mlp1_W, xh, M, NF, HH, mlp1_b, nullptr, nullptr, nullptr, nullptr, 0);

    // z = xh + mask^T @ xh  (reads adj exactly once; xh re-reads hit 192MB L2)
    gin_aggregate<<<256, 256, 0, stream>>>(adj, xh, z);

    for (int l = 0; l < LL; ++l) {
        const float* W1 = gin_lin1_W + (size_t)l * HH * HH;
        const float* W2 = gin_lin2_W + (size_t)l * HH * HH;
        // t = relu(bn1(z @ W1 + b1))
        gemm_rowtile<<<dim3(256, 1), 256, 0, stream>>>(
            z, W1, t, M, HH, HH, gin_lin1_b + l * HH,
            gin_bn1_g + l * HH, gin_bn1_b + l * HH, gin_bn1_m + l * HH, gin_bn1_v + l * HH, 1);
        // h_l = bn2(t @ W2 + b2)  (+relu only for l < L-1)
        gemm_rowtile<<<dim3(256, 1), 256, 0, stream>>>(
            t, W2, (l == 0 ? h0 : h1), M, HH, HH, gin_lin2_b + l * HH,
            gin_bn2_g + l * HH, gin_bn2_b + l * HH, gin_bn2_m + l * HH, gin_bn2_v + l * HH,
            (l == 0) ? 1 : 2);
    }

    colsum_kernel<<<dim3(BB, LL), HH, 0, stream>>>(h0, h1, sums);
    graph_re_kernel<<<BB, HH, 0, stream>>>(sums, mlp2_W, mlp2_b, gre);
    topk_kernel<<<BB, 256, 0, stream>>>(xh, h1, gre, idx);

    tail1_kernel<<<(BB * KK * 512 + 255) / 256, 256, 0, stream>>>(
        idx, emb_table, net_W1, net_b1, hrow);

    // a2 = relu(bn(hrow[32,6144] @ net_W2[6144,512] + b2)) : 2 row tiles x 4 col quadrants
    gemm_rowtile<<<dim3(1, 4), 256, 0, stream>>>(
        hrow, net_W2, a2, BB, KK * 512, 512, net_b2,
        net_bn_g, net_bn_b, net_bn_m, net_bn_v, 1);

    tail3_kernel<<<(BB * TAR + 255) / 256, 256, 0, stream>>>(a2, net_W3, net_b3, (float*)d_out);
}
